// NestedNERModel_22376779612422
// MI455X (gfx1250) — compile-verified, matrix operand order checked by reading, not run on hardware
//
#include <hip/hip_runtime.h>

// ---------------------------------------------------------------------------
// BiLSTM-CRF for MI455X (gfx1250, wave32, WMMA).
// Heavy GEMMs in bf16 WMMA (f32 accum); sequential recurrence as a persistent
// 2-block kernel with LDS-resident bf16 hidden state AND LDS-resident gate
// matrix G (uses CDNA5's 320KB-per-WGP LDS); CRF/Viterbi on VALU.
// ---------------------------------------------------------------------------

typedef __attribute__((ext_vector_type(16))) __bf16 v16bf;
typedef __attribute__((ext_vector_type(8)))  __bf16 v8bf;
typedef __attribute__((ext_vector_type(8)))  float  v8f;

#define B_  64
#define T_  512
#define D_  400
#define Dp_ 416      // D padded to K multiple of 32
#define H_  200
#define Hp_ 224      // H padded to K multiple of 32
#define NG_ 800      // 4*H gates
#define L_  25
#define M_  (B_*T_)  // 32768 rows, (t,b)-major

__device__ __forceinline__ v8f wmma_bf16(v16bf a, v16bf b, v8f c) {
  return __builtin_amdgcn_wmma_f32_16x16x32_bf16(false, a, false, b, (short)0, c,
                                                 false, false);
}

// 16x32 bf16 fragment, CDNA5 per-lane K packing:
// lane l: row = row0 + (l&15), half = l>>4; element e holds
// K = kc + e + (e>=8 ? 8 : 0) + half*8  => two contiguous 8-elem (16B) runs.
__device__ __forceinline__ v16bf load_frag(const __bf16* base, int ld, int row0, int kc) {
  const int lane = threadIdx.x & 31;
  const int r    = row0 + (lane & 15);
  const int half = lane >> 4;
  const __bf16* p = base + (size_t)r * ld + kc + half * 8;
  v8bf lo = *(const v8bf*)(p);
  v8bf hi = *(const v8bf*)(p + 16);
  v16bf o;
#pragma unroll
  for (int i = 0; i < 8; ++i) { o[i] = lo[i]; o[i + 8] = hi[i]; }
  return o;
}

// ---------------------------- conversion kernels ---------------------------

// input_embed (B,T,D) fp32 -> Xbf (T*B, Dp) bf16, (t,b)-major, K zero-padded
__global__ void conv_x_kernel(const float* __restrict__ x, __bf16* __restrict__ dst) {
  size_t i = (size_t)blockIdx.x * blockDim.x + threadIdx.x;
  if (i >= (size_t)M_ * Dp_) return;
  int d = (int)(i % Dp_);
  size_t row = i / Dp_;
  int b = (int)(row % B_);
  int t = (int)(row / B_);
  float v = (d < D_) ? x[((size_t)b * T_ + t) * D_ + d] : 0.f;
  dst[i] = (__bf16)v;
}

// generic fp32 (rows x cols) -> bf16 (rpad x cpad) zero-padded
__global__ void pad_bf16_kernel(const float* __restrict__ src, __bf16* __restrict__ dst,
                                int rows, int cols, int rpad, int cpad) {
  size_t i = (size_t)blockIdx.x * blockDim.x + threadIdx.x;
  if (i >= (size_t)rpad * cpad) return;
  int r = (int)(i / cpad), c = (int)(i % cpad);
  float v = (r < rows && c < cols) ? src[(size_t)r * cols + c] : 0.f;
  dst[i] = (__bf16)v;
}

__global__ void pad_bias_kernel(const float* __restrict__ bl, float* __restrict__ blp) {
  int i = threadIdx.x;
  if (i < 128) blp[i] = (i < L_) ? bl[i] : 0.f;
}

// concat hf,hb (T*B, H) fp32 -> Hcat (T*B, Dp) bf16
__global__ void conv_hcat_kernel(const float* __restrict__ hf, const float* __restrict__ hb,
                                 __bf16* __restrict__ dst) {
  size_t i = (size_t)blockIdx.x * blockDim.x + threadIdx.x;
  if (i >= (size_t)M_ * Dp_) return;
  int d = (int)(i % Dp_);
  size_t row = i / Dp_;
  float v;
  if (d < H_)            v = hf[row * H_ + d];
  else if (d < 2 * H_)   v = hb[row * H_ + (d - H_)];
  else                   v = 0.f;
  dst[i] = (__bf16)v;
}

// ------------------------------- WMMA GEMM ---------------------------------
// C[M x N] = A(M x Kp, bf16) @ W(Npad x Kp, bf16)^T + b1 + b2
// block = 256 thr (8 waves, 2M x 4N), wave tile 32x32, block tile 64x128
__global__ __launch_bounds__(256) void gemm_bias_kernel(
    const __bf16* __restrict__ A, const __bf16* __restrict__ W,
    const float* __restrict__ b1, const float* __restrict__ b2,
    float* __restrict__ C, int M, int N, int Kp, int ldc)
{
  const int wave = threadIdx.x >> 5;
  const int lane = threadIdx.x & 31;
  const int wm = wave >> 2;                    // 0..1
  const int wn = wave & 3;                     // 0..3
  const int m0 = blockIdx.x * 64 + wm * 32;
  const int n0 = blockIdx.y * 128 + wn * 32;

  v8f acc00 = {}, acc01 = {}, acc10 = {}, acc11 = {};
  for (int kc = 0; kc < Kp; kc += 32) {
    if (kc + 32 < Kp)
      __builtin_prefetch(A + (size_t)(m0 + (lane & 15)) * Kp + kc + 32, 0, 3);
    v16bf a0 = load_frag(A, Kp, m0,      kc);
    v16bf a1 = load_frag(A, Kp, m0 + 16, kc);
    v16bf w0 = load_frag(W, Kp, n0,      kc);
    v16bf w1 = load_frag(W, Kp, n0 + 16, kc);
    acc00 = wmma_bf16(a0, w0, acc00);
    acc01 = wmma_bf16(a0, w1, acc01);
    acc10 = wmma_bf16(a1, w0, acc10);
    acc11 = wmma_bf16(a1, w1, acc11);
  }

  const int nn   = lane & 15;
  const int half = lane >> 4;
#pragma unroll
  for (int mt = 0; mt < 2; ++mt) {
#pragma unroll
    for (int nt = 0; nt < 2; ++nt) {
      v8f acc = (mt == 0) ? (nt == 0 ? acc00 : acc01)
                          : (nt == 0 ? acc10 : acc11);
      const int n = n0 + nt * 16 + nn;
      if (n < N) {
        const float bias = (b1 ? b1[n] : 0.f) + (b2 ? b2[n] : 0.f);
#pragma unroll
        for (int r = 0; r < 8; ++r) {
          const int m = m0 + mt * 16 + r + 8 * half;
          C[(size_t)m * ldc + n] = acc[r] + bias;
        }
      }
    }
  }
}

// ---------------------------- persistent LSTM ------------------------------
// grid = 2 (dir), block = 800 thr (25 waves). Per step:
//   G(64x800, LDS) = h_bf(64xHp, LDS) @ Whh^T + P[t]  via WMMA (4x50 tiles)
//   then gate math, mask, state update, write h to hall and LDS (bf16).
// LDS budget: G 200KB + hbf 28KB = 228KB of the WGP's 320KB; one WG per WGP.
__global__ __launch_bounds__(800) void lstm_kernel(
    const float* __restrict__ Pf, const float* __restrict__ Pb,
    const __bf16* __restrict__ Whhf, const __bf16* __restrict__ Whhb,
    const float* __restrict__ mask,
    float* __restrict__ hstf, float* __restrict__ cstf,
    float* __restrict__ hstb, float* __restrict__ cstb,
    float* __restrict__ hallf, float* __restrict__ hallb)
{
  const int dir = blockIdx.x;
  const float*  P    = dir ? Pb   : Pf;
  const __bf16* Whh  = dir ? Whhb : Whhf;
  float* hst  = dir ? hstb  : hstf;
  float* cst  = dir ? cstb  : cstf;
  float* hall = dir ? hallb : hallf;

  __shared__ __attribute__((aligned(16))) __bf16 hbf[B_ * Hp_];   // 28 KB
  __shared__ __attribute__((aligned(16))) float  G[B_ * NG_];     // 200 KB

  const int tid = threadIdx.x;
  for (int i = tid; i < B_ * Hp_; i += 800) hbf[i] = (__bf16)0.f;
  for (int i = tid; i < B_ * H_;  i += 800) { hst[i] = 0.f; cst[i] = 0.f; }
  __syncthreads();

  const int wave = tid >> 5;
  const int lane = tid & 31;
  const int nn   = lane & 15;
  const int half = lane >> 4;

  for (int s = 0; s < T_; ++s) {
    const int t = dir ? (T_ - 1 - s) : s;
    const float* Pt = P + (size_t)t * B_ * NG_;

    // --- step GEMM: 200 16x16 tiles over (4 M-tiles x 50 N-tiles) ---
#pragma unroll 1
    for (int k = 0; k < 8; ++k) {
      const int tileid = wave + 25 * k;       // 0..199
      const int mt = tileid / 50;             // 0..3
      const int nt = tileid % 50;             // 0..49
      const int m0 = mt * 16, n0 = nt * 16;
      v8f acc = {};
#pragma unroll
      for (int kc = 0; kc < Hp_; kc += 32) {
        v16bf a  = load_frag(hbf, Hp_, m0, kc);
        v16bf bw = load_frag(Whh, Hp_, n0, kc);
        acc = wmma_bf16(a, bw, acc);
      }
      const int n = n0 + nn;
#pragma unroll
      for (int r = 0; r < 8; ++r) {
        const int m = m0 + r + 8 * half;
        G[m * NG_ + n] = acc[r] + Pt[(size_t)m * NG_ + n];
      }
    }
    __syncthreads();

    // --- gates: 64*200 = 12800 units, 16 per thread ---
#pragma unroll 1
    for (int k = 0; k < 16; ++k) {
      const int idx = tid + 800 * k;
      const int b = idx / H_;
      const int j = idx % H_;
      const float gi = G[b * NG_ + j];
      const float gf = G[b * NG_ + H_ + j];
      const float gg = G[b * NG_ + 2 * H_ + j];
      const float go = G[b * NG_ + 3 * H_ + j];
      const float iv = 1.f / (1.f + __expf(-gi));
      const float fv = 1.f / (1.f + __expf(-gf));
      const float ov = 1.f / (1.f + __expf(-go));
      const float cp = cst[b * H_ + j];
      const float cn = fv * cp + iv * tanhf(gg);
      const float hn = ov * tanhf(cn);
      const float m  = mask[(size_t)b * T_ + t];
      const float h  = m * hn + (1.f - m) * hst[b * H_ + j];
      const float c  = m * cn + (1.f - m) * cp;
      hst[b * H_ + j] = h;
      cst[b * H_ + j] = c;
      hall[((size_t)t * B_ + b) * H_ + j] = h;
      hbf[b * Hp_ + j] = (__bf16)h;
    }
    __syncthreads();
  }
}

// ---------------------------- emissions mask -------------------------------
__global__ void mask_e_kernel(float* __restrict__ e, const float* __restrict__ mask) {
  size_t i = (size_t)blockIdx.x * blockDim.x + threadIdx.x;
  if (i >= (size_t)M_ * 32) return;
  size_t row = i / 32;
  int b = (int)(row % B_);
  int t = (int)(row / B_);
  e[i] *= mask[(size_t)b * T_ + t];
}

// ------------------------------- CRF llh -----------------------------------
// grid = B, block = 32. alpha in LDS; thread j owns label j.
__global__ __launch_bounds__(32) void crf_kernel(
    const float* __restrict__ e, const float* __restrict__ mask,
    const int* __restrict__ labels, const float* __restrict__ trans,
    const float* __restrict__ start, const float* __restrict__ endv,
    float* __restrict__ llh)
{
  const int b = blockIdx.x;
  const int j = threadIdx.x;
  __shared__ float alpha[L_], alnew[L_], tr[L_ * L_];
  for (int i = j; i < L_ * L_; i += 32) tr[i] = trans[i];
  if (j < L_) alpha[j] = start[j] + e[(size_t)b * 32 + j];
  __syncthreads();

  for (int t = 1; t < T_; ++t) {
    const float m = mask[(size_t)b * T_ + t];
    if (j < L_) {
      float nv;
      if (m > 0.f) {
        float mx = -3.4e38f;
        for (int i = 0; i < L_; ++i) mx = fmaxf(mx, alpha[i] + tr[i * L_ + j]);
        float sum = 0.f;
        for (int i = 0; i < L_; ++i) sum += __expf(alpha[i] + tr[i * L_ + j] - mx);
        nv = mx + __logf(sum) + e[((size_t)t * B_ + b) * 32 + j];
      } else {
        nv = alpha[j];
      }
      alnew[j] = nv;
    }
    __syncthreads();
    if (j < L_) alpha[j] = alnew[j];
    __syncthreads();
  }

  if (j == 0) {
    float mx = -3.4e38f;
    for (int i = 0; i < L_; ++i) mx = fmaxf(mx, alpha[i] + endv[i]);
    float sum = 0.f;
    for (int i = 0; i < L_; ++i) sum += __expf(alpha[i] + endv[i] - mx);
    const float logZ = mx + __logf(sum);

    const int* lab = labels + (size_t)b * T_;
    float num = start[lab[0]] + e[(size_t)b * 32 + lab[0]];
    int last = 0;
    for (int t = 0; t < T_; ++t) if (mask[(size_t)b * T_ + t] > 0.f) last = t;
    for (int t = 1; t < T_; ++t) {
      const float m = mask[(size_t)b * T_ + t];
      num += m * (e[((size_t)t * B_ + b) * 32 + lab[t]] + trans[lab[t - 1] * L_ + lab[t]]);
    }
    num += endv[lab[last]];
    llh[b] = num - logZ;
  }
}

// -------------------------------- Viterbi ----------------------------------
__global__ __launch_bounds__(32) void viterbi_kernel(
    const float* __restrict__ e, const float* __restrict__ mask,
    const float* __restrict__ trans, const float* __restrict__ start,
    const float* __restrict__ endv, unsigned char* __restrict__ ptrs,
    float* __restrict__ preds)
{
  const int b = blockIdx.x;
  const int j = threadIdx.x;
  __shared__ float delta[L_], dnew[L_], tr[L_ * L_];
  for (int i = j; i < L_ * L_; i += 32) tr[i] = trans[i];
  if (j < L_) delta[j] = start[j] + e[(size_t)b * 32 + j];
  __syncthreads();

  for (int t = 1; t < T_; ++t) {
    const float m = mask[(size_t)b * T_ + t];
    if (j < L_) {
      int   bi;
      float bv;
      if (m > 0.f) {
        bv = -3.4e38f; bi = 0;
        for (int i = 0; i < L_; ++i) {
          const float v = delta[i] + tr[i * L_ + j];
          if (v > bv) { bv = v; bi = i; }
        }
        bv += e[((size_t)t * B_ + b) * 32 + j];
      } else {
        bv = delta[j]; bi = j;
      }
      dnew[j] = bv;
      ptrs[((size_t)(t - 1) * B_ + b) * L_ + j] = (unsigned char)bi;
    }
    __syncthreads();
    if (j < L_) delta[j] = dnew[j];
    __syncthreads();
  }

  if (j == 0) {
    int y = 0; float bv = -3.4e38f;
    for (int i = 0; i < L_; ++i) {
      const float v = delta[i] + endv[i];
      if (v > bv) { bv = v; y = i; }
    }
    preds[(size_t)b * T_ + (T_ - 1)] =
        (mask[(size_t)b * T_ + (T_ - 1)] > 0.f) ? (float)y : 0.f;
    for (int t = T_ - 1; t >= 1; --t) {
      y = ptrs[((size_t)(t - 1) * B_ + b) * L_ + y];
      preds[(size_t)b * T_ + (t - 1)] =
          (mask[(size_t)b * T_ + (t - 1)] > 0.f) ? (float)y : 0.f;
    }
  }
}

__global__ void loss_kernel(const float* __restrict__ llh, float* __restrict__ out) {
  __shared__ float s[B_];
  const int i = threadIdx.x;
  s[i] = llh[i];
  __syncthreads();
  if (i == 0) {
    float a = 0.f;
    for (int k = 0; k < B_; ++k) a += s[k];
    out[0] = -a / (float)B_;
  }
}

// ------------------------------- launcher ----------------------------------
extern "C" void kernel_launch(void* const* d_in, const int* in_sizes, int n_in,
                              void* d_out, int out_size, void* d_ws, size_t ws_size,
                              hipStream_t stream) {
  const float* x      = (const float*)d_in[0];
  const float* mask   = (const float*)d_in[1];
  const int*   labels = (const int*)d_in[2];
  const float* Wih_f  = (const float*)d_in[3];
  const float* Whh_f  = (const float*)d_in[4];
  const float* bih_f  = (const float*)d_in[5];
  const float* bhh_f  = (const float*)d_in[6];
  const float* Wih_b  = (const float*)d_in[7];
  const float* Whh_b  = (const float*)d_in[8];
  const float* bih_b  = (const float*)d_in[9];
  const float* bhh_b  = (const float*)d_in[10];
  const float* Wl     = (const float*)d_in[11];
  const float* bl     = (const float*)d_in[12];
  const float* trans  = (const float*)d_in[13];
  const float* start  = (const float*)d_in[14];
  const float* endv   = (const float*)d_in[15];
  float* out = (float*)d_out;

  char* p = (char*)d_ws;
  auto carve = [&](size_t bytes) -> char* {
    char* q = p;
    p += (bytes + 255) & ~(size_t)255;
    return q;
  };

  __bf16* Xbf   = (__bf16*)carve((size_t)M_ * Dp_ * 2);
  __bf16* Wihf  = (__bf16*)carve((size_t)896 * Dp_ * 2);   // 800 -> 896 (N grid pad)
  __bf16* Wihb  = (__bf16*)carve((size_t)896 * Dp_ * 2);
  __bf16* Whhf  = (__bf16*)carve((size_t)NG_ * Hp_ * 2);
  __bf16* Whhb  = (__bf16*)carve((size_t)NG_ * Hp_ * 2);
  __bf16* Wlp   = (__bf16*)carve((size_t)128 * Dp_ * 2);   // 25 -> 128 rows
  float*  blp   = (float*)carve(128 * 4);
  float*  Pf    = (float*)carve((size_t)M_ * NG_ * 4);
  float*  Pb    = (float*)carve((size_t)M_ * NG_ * 4);
  float*  hstf  = (float*)carve((size_t)B_ * H_ * 4);
  float*  cstf  = (float*)carve((size_t)B_ * H_ * 4);
  float*  hstb  = (float*)carve((size_t)B_ * H_ * 4);
  float*  cstb  = (float*)carve((size_t)B_ * H_ * 4);
  float*  hallf = (float*)carve((size_t)M_ * H_ * 4);
  float*  hallb = (float*)carve((size_t)M_ * H_ * 4);
  __bf16* Hcat  = (__bf16*)carve((size_t)M_ * Dp_ * 2);
  float*  eM    = (float*)carve((size_t)M_ * 32 * 4);
  float*  llh   = (float*)carve(B_ * 4);
  unsigned char* ptrs = (unsigned char*)carve((size_t)(T_ - 1) * B_ * L_);

  const size_t nxd = (size_t)M_ * Dp_;

  conv_x_kernel<<<(unsigned)((nxd + 255) / 256), 256, 0, stream>>>(x, Xbf);
  pad_bf16_kernel<<<(896 * Dp_ + 255) / 256, 256, 0, stream>>>(Wih_f, Wihf, NG_, D_, 896, Dp_);
  pad_bf16_kernel<<<(896 * Dp_ + 255) / 256, 256, 0, stream>>>(Wih_b, Wihb, NG_, D_, 896, Dp_);
  pad_bf16_kernel<<<(NG_ * Hp_ + 255) / 256, 256, 0, stream>>>(Whh_f, Whhf, NG_, H_, NG_, Hp_);
  pad_bf16_kernel<<<(NG_ * Hp_ + 255) / 256, 256, 0, stream>>>(Whh_b, Whhb, NG_, H_, NG_, Hp_);
  pad_bf16_kernel<<<(128 * Dp_ + 255) / 256, 256, 0, stream>>>(Wl, Wlp, L_, D_, 128, Dp_);
  pad_bias_kernel<<<1, 128, 0, stream>>>(bl, blp);

  // Input projections: (32768 x 416) @ (800 x 416)^T + biases
  gemm_bias_kernel<<<dim3(M_ / 64, 7), 256, 0, stream>>>(Xbf, Wihf, bih_f, bhh_f,
                                                         Pf, M_, NG_, Dp_, NG_);
  gemm_bias_kernel<<<dim3(M_ / 64, 7), 256, 0, stream>>>(Xbf, Wihb, bih_b, bhh_b,
                                                         Pb, M_, NG_, Dp_, NG_);

  // Persistent bidirectional recurrence (block 0 = forward, block 1 = backward)
  lstm_kernel<<<2, 800, 0, stream>>>(Pf, Pb, Whhf, Whhb, mask,
                                     hstf, cstf, hstb, cstb, hallf, hallb);

  conv_hcat_kernel<<<(unsigned)((nxd + 255) / 256), 256, 0, stream>>>(hallf, hallb, Hcat);

  // Emissions: (32768 x 416) @ (25->32 x 416)^T + bl, then mask
  gemm_bias_kernel<<<dim3(M_ / 64, 1), 256, 0, stream>>>(Hcat, Wlp, blp, nullptr,
                                                         eM, M_, 32, Dp_, 32);
  mask_e_kernel<<<(M_ * 32 + 255) / 256, 256, 0, stream>>>(eM, mask);

  crf_kernel<<<B_, 32, 0, stream>>>(eM, mask, labels, trans, start, endv, llh);
  viterbi_kernel<<<B_, 32, 0, stream>>>(eM, mask, trans, start, endv, ptrs, out + 1);
  loss_kernel<<<1, B_, 0, stream>>>(llh, out);
}
